// SimpleSNN_50689204027532
// MI455X (gfx1250) — matrix operand order repaired and checked by reading, not compiled, not run on hardware
//
#include <hip/hip_runtime.h>

typedef __attribute__((ext_vector_type(16))) _Float16 v16h;
typedef __attribute__((ext_vector_type(8)))  _Float16 v8h;
typedef __attribute__((ext_vector_type(8)))  float    v8f;
typedef __attribute__((ext_vector_type(2)))  float    v2f;

// reference problem sizes
constexpr int B_   = 131072;
constexpr int T_   = 10;
constexpr int IN_  = 50;
constexpr int HID_ = 100;
constexpr int OUT_ = 10;

constexpr int KPAD  = 64;    // IN padded -> two K=32 WMMA steps
constexpr int NPAD  = 112;   // HID padded -> 7 N-tiles of 16
constexpr int NT_H  = 7;     // hid tiles
constexpr int NT_I  = 4;     // in tiles (50 -> 64)
constexpr int BTILE = 128;   // batch rows per block (8 waves x 16-row M-tile)

// LDS layout (phase-overlapped), all offsets 32B aligned
constexpr int W2T_OFF = 0;                  // float [100][10]  = 4000 B
constexpr int W1T_OFF = 4000;               // f16  [64][112]   = 14336 B
constexpr int SPK_OFF = W1T_OFF + 14336;    // f16  [128][64]   = 16384 B
constexpr int ST_OFF  = SPK_OFF + 16384;    // f16  [112][128]  = 28672 B
constexpr int SMEM_BYTES = ST_OFF + 28672;  // 63392 B
constexpr int SUM_OFF = 4000;               // float [128][100] (readout phase, overlaps w1t/spk/sT)

__global__ __launch_bounds__(256) void snn_fused(
    const float* __restrict__ x_static, const float* __restrict__ rand_u,
    const float* __restrict__ W1, const float* __restrict__ b1,
    const float* __restrict__ W2, const float* __restrict__ b2,
    float* __restrict__ out, float* __restrict__ trace)
{
  __shared__ alignas(32) char smem[SMEM_BYTES];
  float*    w2t = (float*)   (smem + W2T_OFF);
  _Float16* w1t = (_Float16*)(smem + W1T_OFF);
  _Float16* spk = (_Float16*)(smem + SPK_OFF);
  _Float16* sT  = (_Float16*)(smem + ST_OFF);

  const int tid  = threadIdx.x;
  const int w    = tid >> 5;           // wave id 0..7
  const int l    = tid & 31;           // lane
  const int lh   = l >> 4;             // half-wave select
  const int ll   = l & 15;
  const int mrow = w * 16;             // wave's local batch-row base
  const int rowBase = blockIdx.x * BTILE;
  const int koffA = lh ? 8 : 0;        // A-fragment K sub-offset (ISA 16-bit A layout)

  // one-time LDS fills: W1^T (f16, zero-padded) and W2^T (f32)
  for (int idx = tid; idx < KPAD * NPAD; idx += 256) {
    int k = idx / NPAD, n = idx % NPAD;
    w1t[idx] = (_Float16)((k < IN_ && n < HID_) ? W1[n * IN_ + k] : 0.0f);
  }
  for (int idx = tid; idx < HID_ * OUT_; idx += 256) {
    int k = idx / OUT_, o = idx % OUT_;
    w2t[idx] = W2[o * HID_ + k];
  }

  // per-lane bias cache (C-layout column n = nt*16 + ll)
  float biasv[NT_H];
  #pragma unroll
  for (int nt = 0; nt < NT_H; ++nt) {
    int n = nt * 16 + ll;
    biasv[nt] = (n < HID_) ? b1[n] : 0.0f;
  }

  const v8f vzero = {};
  v8f vmem[NT_H], ssum[NT_H];        // membrane potential, spike counts
  #pragma unroll
  for (int nt = 0; nt < NT_H; ++nt) { vmem[nt] = vzero; ssum[nt] = vzero; }

  v8f tracc[4];                      // trace jobs j = w + 8*ji, j < 28
  #pragma unroll
  for (int ji = 0; ji < 4; ++ji) tracc[ji] = vzero;

  #pragma unroll 1
  for (int t = 0; t < T_; ++t) {
    // ---- Poisson/Bernoulli encode into LDS (f16 {0,1}), KPAD zero pad ----
    {
      const size_t ubase = (size_t)t * B_ * IN_;
      #pragma unroll
      for (int it8 = 0; it8 < (BTILE * KPAD) / 256; ++it8) {
        int idx = it8 * 256 + tid;
        int r = idx >> 6, k = idx & 63;
        float s = 0.0f;
        if (k < IN_) {
          size_t g = (size_t)(rowBase + r) * IN_ + k;
          s = (rand_u[ubase + g] < x_static[g]) ? 1.0f : 0.0f;
        }
        spk[idx] = (_Float16)s;
      }
      if (t + 1 < T_) {  // prefetch next timestep's contiguous rand_u slice
        const char* nb = (const char*)(rand_u + (size_t)(t + 1) * B_ * IN_
                                              + (size_t)rowBase * IN_);
        int off = tid * 128;
        if (off < BTILE * IN_ * 4) __builtin_prefetch(nb + off, 0, 1);
      }
    }
    __syncthreads();

    // ---- layer1 GEMM (WMMA f16) + LIF update + spike, write s^T to LDS ----
    #pragma unroll
    for (int nt = 0; nt < NT_H; ++nt) {
      v8f c = vzero;
      #pragma unroll
      for (int kt = 0; kt < 2; ++kt) {
        const _Float16* ap = spk + (mrow + ll) * KPAD + kt * 32 + koffA;
        v8h alo = *(const v8h*)ap;
        v8h ahi = *(const v8h*)(ap + 16);
        v16h a;
        #pragma unroll
        for (int i = 0; i < 8; ++i) { a[i] = alo[i]; a[i + 8] = ahi[i]; }
        v16h bm;
        #pragma unroll
        for (int i = 0; i < 8; ++i) {
          bm[i]     = w1t[(kt * 32 + koffA + i) * NPAD + nt * 16 + ll];
          bm[i + 8] = w1t[(kt * 32 + 16 + koffA + i) * NPAD + nt * 16 + ll];
        }
        c = __builtin_amdgcn_wmma_f32_16x16x32_f16(false, a, false, bm,
                                                   (short)0, c, false, false);
      }
      #pragma unroll
      for (int r = 0; r < 8; ++r) {
        float vn = 0.9f * vmem[nt][r] + 0.1f * (c[r] + biasv[nt]); // v + dt*tau*(i-v)
        float s  = (vn > 1.0f) ? 1.0f : 0.0f;                      // fire (v_th=1)
        vmem[nt][r] = vn * (1.0f - s);                             // reset
        ssum[nt][r] += s;
        sT[(nt * 16 + ll) * BTILE + (mrow + r + 8 * lh)] = (_Float16)s;
      }
    }
    __syncthreads();

    // ---- eligibility trace: tr = 0.95*tr + s^T x spikes (WMMA f16) ----
    #pragma unroll
    for (int ji = 0; ji < 4; ++ji) {
      if (w + 8 * ji < NT_H * NT_I) {    // wave-uniform guard
        int j = w + 8 * ji;
        int nth = j >> 2, iti = j & 3;
        v8f tr = tracc[ji];
        #pragma unroll
        for (int r = 0; r < 8; ++r) tr[r] *= 0.95f;
        #pragma unroll
        for (int kt = 0; kt < 4; ++kt) {
          const _Float16* ap = sT + (nth * 16 + ll) * BTILE + kt * 32 + koffA;
          v8h alo = *(const v8h*)ap;
          v8h ahi = *(const v8h*)(ap + 16);
          v16h a;
          #pragma unroll
          for (int i = 0; i < 8; ++i) { a[i] = alo[i]; a[i + 8] = ahi[i]; }
          v16h bm;
          #pragma unroll
          for (int i = 0; i < 8; ++i) {
            bm[i]     = spk[(kt * 32 + koffA + i) * KPAD + iti * 16 + ll];
            bm[i + 8] = spk[(kt * 32 + 16 + koffA + i) * KPAD + iti * 16 + ll];
          }
          tr = __builtin_amdgcn_wmma_f32_16x16x32_f16(false, a, false, bm,
                                                      (short)0, tr, false, false);
        }
        tracc[ji] = tr;
      }
    }
    __syncthreads();  // spk/sT rewritten next timestep
  }

  // ---- trace writeback: /B and atomic-accumulate across blocks ----
  const float invB = 1.0f / (float)B_;
  #pragma unroll
  for (int ji = 0; ji < 4; ++ji) {
    if (w + 8 * ji < NT_H * NT_I) {
      int j = w + 8 * ji;
      int nth = j >> 2, iti = j & 3;
      #pragma unroll
      for (int r = 0; r < 8; ++r) {
        int hid = nth * 16 + r + 8 * lh;
        int inn = iti * 16 + ll;
        if (hid < HID_ && inn < IN_)
          atomicAdd(&trace[hid * IN_ + inn], tracc[ji][r] * invB);
      }
    }
  }

  // ---- stash spike counts in LDS (overlaps dead spk/sT/w1t) ----
  float* sums = (float*)(smem + SUM_OFF);  // [128][100]
  #pragma unroll
  for (int nt = 0; nt < NT_H; ++nt) {
    int n = nt * 16 + ll;
    if (n < HID_) {
      #pragma unroll
      for (int r = 0; r < 8; ++r)
        sums[(mrow + r + 8 * lh) * HID_ + n] = ssum[nt][r];
    }
  }
  __syncthreads();

  // ---- readout: out = summed x W2^T + b2 ----
#if __has_builtin(__builtin_amdgcn_wmma_f32_16x16x4_f32)
  {
    v8f acc = vzero;
    const int m = mrow + ll;
    #pragma unroll
    for (int k0 = 0; k0 < HID_; k0 += 4) {
      int kA = k0 + 2 * lh;
      v2f a; a.x = sums[m * HID_ + kA]; a.y = sums[m * HID_ + kA + 1];
      v2f bm;
      bm.x = (ll < OUT_) ? w2t[kA * OUT_ + ll] : 0.0f;
      bm.y = (ll < OUT_) ? w2t[(kA + 1) * OUT_ + ll] : 0.0f;
      acc = __builtin_amdgcn_wmma_f32_16x16x4_f32(false, a, false, bm,
                                                  (short)0, acc, false, false);
    }
    #pragma unroll
    for (int r = 0; r < 8; ++r) {
      int o = ll, mm = mrow + r + 8 * lh;
      if (o < OUT_) out[(size_t)(rowBase + mm) * OUT_ + o] = acc[r] + b2[o];
    }
  }
#else
  for (int q = l; q < 16 * OUT_; q += 32) {
    int mm = mrow + q / OUT_, o = q % OUT_;
    float a = b2[o];
    for (int h = 0; h < HID_; ++h) a += sums[mm * HID_ + h] * W2[o * HID_ + h];
    out[(size_t)(rowBase + mm) * OUT_ + o] = a;
  }
#endif
}

__global__ void zero_f32(float* __restrict__ p, int n) {
  int i = blockIdx.x * blockDim.x + threadIdx.x;
  if (i < n) p[i] = 0.0f;
}

extern "C" void kernel_launch(void* const* d_in, const int* in_sizes, int n_in,
                              void* d_out, int out_size, void* d_ws, size_t ws_size,
                              hipStream_t stream) {
  (void)in_sizes; (void)n_in; (void)out_size; (void)d_ws; (void)ws_size;
  const float* x_static = (const float*)d_in[0];
  const float* rand_u   = (const float*)d_in[1];
  const float* W1       = (const float*)d_in[2];
  const float* b1       = (const float*)d_in[3];
  const float* W2       = (const float*)d_in[4];
  const float* b2       = (const float*)d_in[5];

  float* out   = (float*)d_out;
  float* trace = out + (size_t)B_ * OUT_;   // outputs concatenated: out then trace

  zero_f32<<<(HID_ * IN_ + 255) / 256, 256, 0, stream>>>(trace, HID_ * IN_);
  snn_fused<<<B_ / BTILE, 256, 0, stream>>>(x_static, rand_u, W1, b1, W2, b2,
                                            out, trace);
}
